// EstVAEStudent_73143293051311
// MI455X (gfx1250) — matrix operand order repaired and checked by reading, not compiled, not run on hardware
//
#include <hip/hip_runtime.h>
#include <stdint.h>
#include <stddef.h>

// ---------------- model dims ----------------
#define BN      4096
#define CUR_OBS 256
#define HIST_C  96
#define HIST_T  25
#define FUT_FLAT 2560
#define HID     1024
#define PROJ    512
#define HLAT    64
#define FLATD   64
#define CONV1C  256
#define CONV2C  512
#define CONV3C  1024
#define NE      16
#define OUTD    23

// ---------------- wmma types ----------------
typedef __attribute__((ext_vector_type(16))) __bf16 v16bf;
typedef __attribute__((ext_vector_type(8)))  float  v8f;
typedef __attribute__((ext_vector_type(8)))  unsigned short ush8;
typedef __attribute__((ext_vector_type(4)))  int    v4i_;

__device__ __forceinline__ unsigned short f32_to_bf16(float f) {
  uint32_t u = __builtin_bit_cast(uint32_t, f);
  u += 0x7FFFu + ((u >> 16) & 1u);            // round-to-nearest-even
  return (unsigned short)(u >> 16);
}
__device__ __forceinline__ float bf16_to_f32(unsigned short h) {
  uint32_t u = ((uint32_t)h) << 16;
  return __builtin_bit_cast(float, u);
}

union FragU { ush8 h[2]; v16bf v; };

// A fragment (16x32, M x K): lane row = lane%16; K chunks at (lane/16)*8 and 16+(lane/16)*8
__device__ __forceinline__ v16bf frag_a(const unsigned short* p /*row base at k0*/, int lane) {
  const int kb = (lane >> 4) * 8;
  FragU u;
  u.h[0] = *(const ush8*)(p + kb);
  u.h[1] = *(const ush8*)(p + kb + 16);
  return u.v;
}
// B fragment (32x16, K x N) from B^T storage [N][K]: lane col = lane%16; K chunks at (lane/16)*8 and 16+(lane/16)*8
__device__ __forceinline__ v16bf frag_b(const unsigned short* p /*col base at k0*/, int lane) {
  const int kb = (lane >> 4) * 8;
  FragU u;
  u.h[0] = *(const ush8*)(p + kb);
  u.h[1] = *(const ush8*)(p + kb + 16);
  return u.v;
}
__device__ __forceinline__ v8f wmma_bf16(v16bf a, v16bf b, v8f c) {
  return __builtin_amdgcn_wmma_f32_16x16x32_bf16(false, a, false, b, (short)0, c, false, false);
}
__device__ __forceinline__ float silu_f(float v) { return v / (1.f + __expf(-v)); }

// ---------------- CDNA5 async global->LDS copy (ASYNCcnt path), with fallback ----------------
#if defined(__gfx1250__) && __has_builtin(__builtin_amdgcn_global_load_async_to_lds_b128) && \
    __has_builtin(__builtin_amdgcn_s_wait_asynccnt)
#define HAVE_ASYNC_LDS 1
#else
#define HAVE_ASYNC_LDS 0
#endif

__device__ __forceinline__ void cp16_g2l(unsigned short* ldst, const unsigned short* g) {
#if HAVE_ASYNC_LDS
  __builtin_amdgcn_global_load_async_to_lds_b128(
      (__attribute__((address_space(1))) v4i_*)g,
      (__attribute__((address_space(3))) v4i_*)ldst, 0, 0);
#else
  *(ush8*)ldst = *(const ush8*)g;
#endif
}
__device__ __forceinline__ void async_fence() {
#if HAVE_ASYNC_LDS
  __builtin_amdgcn_s_wait_asynccnt(0);
#endif
}

// ---------------- weight prep kernels ----------------
// src fp32 [batch][K][N] -> dst bf16 [batch][N][K]  (transpose + convert)
__global__ void k_transpose_bf16(const float* __restrict__ src, unsigned short* __restrict__ dst,
                                 int K, int N) {
  const size_t total = (size_t)K * N;
  const size_t base  = (size_t)blockIdx.y * total;
  for (size_t i = (size_t)blockIdx.x * blockDim.x + threadIdx.x; i < total;
       i += (size_t)gridDim.x * blockDim.x) {
    size_t n = i / (size_t)K;
    size_t k = i - n * (size_t)K;
    dst[base + i] = f32_to_bf16(src[base + k * (size_t)N + n]);
  }
}
// flat fp32 -> bf16
__global__ void k_convert_bf16(const float* __restrict__ src, unsigned short* __restrict__ dst, size_t n) {
  for (size_t i = (size_t)blockIdx.x * blockDim.x + threadIdx.x; i < n;
       i += (size_t)gridDim.x * blockDim.x)
    dst[i] = f32_to_bf16(src[i]);
}

// ---------------- generic WMMA GEMM ----------------
// O = epilogue(A[M,K] @ (Bt[N,K])^T + bias)
// Requirements (guaranteed by all call sites): M % 128 == 0, K % 32 == 0.
struct GemmParams {
  const void* A;  int a_dtype;  // 0=f32, 1=bf16
  int a_mode;                   // 0=dense, 1=im2col over [B,T,C] f32, 2=im2col over [B,C,T] bf16
  long lda;
  int cin, tin, tout, cstride;  // im2col
  const unsigned short* Bt;     // bf16 [N][K]
  const float* bias;            // [N] or null
  const unsigned short* aux;    // bf16 [M,N] (stride ldo) for ep==2
  void* O; int o_dtype;         // 0=f32, 1=bf16
  int o_mode;                   // 0=dense (ldo), 1=CHW: [B, N, tout]
  long ldo;
  int M, N, K;
  int ep;                       // 0 none, 1 silu, 2 out = (acc+bias)*aux
};

#define ALD 40   // padded LDS row stride (halfwords): conflict-free 16B lane reads

// 128x128 workgroup tile, 8 wave32s in 4x2, each wave computes 32x64 via 2x4 wmma frags.
__global__ __launch_bounds__(256) void k_gemm(GemmParams p) {
  __shared__ unsigned short As[128 * ALD];
  __shared__ unsigned short Bs[128 * ALD];
  const int tid  = threadIdx.x;
  const int lane = tid & 31;
  const int wave = tid >> 5;
  const int wm2 = wave & 3;   // 32-row block
  const int wn2 = wave >> 2;  // 64-col block
  const int row0 = blockIdx.y * 128;
  const int col0 = blockIdx.x * 128;
  const int srow = tid >> 2;          // 0..63 : staging row (two rows/thread)
  const int kc   = (tid & 3) * 8;     // 0,8,16,24 : staging k offset

  const bool aFast = (p.a_mode == 0);
  const bool bFast = (col0 + 128 <= p.N);

  v8f acc[2][4] = {};

  for (int k0 = 0; k0 < p.K; k0 += 32) {
    bool used_async = false;
    // ---- stage A tile (128 x 32) as bf16 ----
    if (aFast) {
      if (p.a_dtype == 1) {
        const unsigned short* a = (const unsigned short*)p.A;
        cp16_g2l(&As[srow * ALD + kc],        a + (size_t)(row0 + srow) * p.lda + k0 + kc);
        cp16_g2l(&As[(srow + 64) * ALD + kc], a + (size_t)(row0 + srow + 64) * p.lda + k0 + kc);
        used_async = true;
      } else {
        const float* a = (const float*)p.A;
        for (int q = 0; q < 2; ++q) {
          const int r = srow + q * 64;
          const float* src = a + (size_t)(row0 + r) * p.lda + k0 + kc;
          const float4 f0 = *(const float4*)(src);
          const float4 f1 = *(const float4*)(src + 4);
          union { unsigned short s[8]; ush8 v; } o;
          o.s[0] = f32_to_bf16(f0.x); o.s[1] = f32_to_bf16(f0.y);
          o.s[2] = f32_to_bf16(f0.z); o.s[3] = f32_to_bf16(f0.w);
          o.s[4] = f32_to_bf16(f1.x); o.s[5] = f32_to_bf16(f1.y);
          o.s[6] = f32_to_bf16(f1.z); o.s[7] = f32_to_bf16(f1.w);
          *(ush8*)&As[r * ALD + kc] = o.v;
        }
      }
    } else {
      // im2col gather (conv as GEMM): pad=1, kernel=3
      for (int q = 0; q < 2; ++q) {
        const int mr = srow + q * 64;
        const int r = row0 + mr;
        const int b = r / p.tout;
        const int t = r - b * p.tout;
        unsigned short* dst = &As[mr * ALD + kc];
        for (int j = 0; j < 8; ++j) {
          const int k = k0 + kc + j;
          unsigned short hv = 0;
          if (r < p.M) {
            const int ci = k / 3;
            const int kk = k - ci * 3;
            const int ti = t * p.cstride + kk - 1;
            if (ti >= 0 && ti < p.tin) {
              if (p.a_mode == 1)
                hv = f32_to_bf16(((const float*)p.A)[((size_t)b * p.tin + ti) * p.cin + ci]);
              else
                hv = ((const unsigned short*)p.A)[((size_t)b * p.cin + ci) * p.tin + ti];
            }
          }
          dst[j] = hv;
        }
      }
    }
    // ---- stage B tile (128 cols x 32 k) from Bt[N][K] ----
    if (bFast) {
      cp16_g2l(&Bs[srow * ALD + kc],        p.Bt + (size_t)(col0 + srow) * p.K + k0 + kc);
      cp16_g2l(&Bs[(srow + 64) * ALD + kc], p.Bt + (size_t)(col0 + srow + 64) * p.K + k0 + kc);
      used_async = true;
    } else {
      for (int q = 0; q < 2; ++q) {
        const int nr = srow + q * 64;
        const int n = col0 + nr;
        unsigned short* dst = &Bs[nr * ALD + kc];
        for (int j = 0; j < 8; ++j) {
          const int k = k0 + kc + j;
          dst[j] = (n < p.N) ? p.Bt[(size_t)n * p.K + k] : (unsigned short)0;
        }
      }
    }
    if (used_async) async_fence();
    __syncthreads();

    const v16bf a0 = frag_a(&As[(wm2 * 32 +      (lane & 15)) * ALD], lane);
    const v16bf a1 = frag_a(&As[(wm2 * 32 + 16 + (lane & 15)) * ALD], lane);
#pragma unroll
    for (int jb = 0; jb < 4; ++jb) {
      const v16bf b = frag_b(&Bs[(wn2 * 64 + jb * 16 + (lane & 15)) * ALD], lane);
      acc[0][jb] = wmma_bf16(a0, b, acc[0][jb]);
      acc[1][jb] = wmma_bf16(a1, b, acc[1][jb]);
    }
    __syncthreads();
  }

  // ---- epilogue ----
  const int nloc = lane & 15;
  const int mofs = (lane >= 16) ? 8 : 0;
#pragma unroll
  for (int ib = 0; ib < 2; ++ib) {
#pragma unroll
    for (int jb = 0; jb < 4; ++jb) {
      const int c = col0 + wn2 * 64 + jb * 16 + nloc;
      if (c >= p.N) continue;
      const float bias = p.bias ? p.bias[c] : 0.f;
      for (int i = 0; i < 8; ++i) {
        const int r = row0 + wm2 * 32 + ib * 16 + i + mofs;
        if (r >= p.M) continue;
        float v = acc[ib][jb][i] + bias;
        if (p.ep == 1)      v = silu_f(v);
        else if (p.ep == 2) v *= bf16_to_f32(p.aux[(size_t)r * p.ldo + c]);
        if (p.o_mode == 0) {
          if (p.o_dtype == 0) ((float*)p.O)[(size_t)r * p.ldo + c] = v;
          else ((unsigned short*)p.O)[(size_t)r * p.ldo + c] = f32_to_bf16(v);
        } else { // CHW output for conv chains: [B, N, tout]
          const int b = r / p.tout;
          const int t = r - b * p.tout;
          ((unsigned short*)p.O)[((size_t)b * p.N + c) * p.tout + t] = f32_to_bf16(v);
        }
      }
    }
  }
}

// ---------------- small kernels ----------------
__global__ void k_meanpool7(const unsigned short* __restrict__ h3, unsigned short* __restrict__ out) {
  const size_t total = (size_t)BN * CONV3C;
  for (size_t i = (size_t)blockIdx.x * blockDim.x + threadIdx.x; i < total;
       i += (size_t)gridDim.x * blockDim.x) {
    float s = 0.f;
    const unsigned short* p = h3 + i * 7;
    for (int t = 0; t < 7; ++t) s += bf16_to_f32(p[t]);
    out[i] = f32_to_bf16(s * (1.f / 7.f));
  }
}

// Build A_cat[B,384] bf16 = [cur_obs | hist_lat | z],  z = mu + exp(0.5*logvar)*noise
__global__ void k_reparam_concat(const float* __restrict__ cur_obs,
                                 const unsigned short* __restrict__ hist_lat,
                                 const float* __restrict__ e3,      // [B,128]: mu | logvar
                                 const float* __restrict__ noise,   // [B,64]
                                 unsigned short* __restrict__ acat) {
  const size_t total = (size_t)BN * 384;
  for (size_t i = (size_t)blockIdx.x * blockDim.x + threadIdx.x; i < total;
       i += (size_t)gridDim.x * blockDim.x) {
    const size_t b = i / 384;
    const int    c = (int)(i - b * 384);
    float v;
    if (c < CUR_OBS) v = cur_obs[b * CUR_OBS + c];
    else if (c < CUR_OBS + HLAT) v = bf16_to_f32(hist_lat[b * HLAT + (c - CUR_OBS)]);
    else {
      const int j = c - CUR_OBS - HLAT;
      const float mu = e3[b * 128 + j];
      const float lv = e3[b * 128 + 64 + j];
      v = mu + __expf(0.5f * lv) * noise[b * FLATD + j];
    }
    acat[i] = f32_to_bf16(v);
  }
}

// DeepseekV3 sigmoid + group-top2 + top4 router (fp32 scalar, one thread / row)
__global__ void k_router_topk(const float* __restrict__ logits, float* __restrict__ w) {
  const int b = blockIdx.x * blockDim.x + threadIdx.x;
  if (b >= BN) return;
  float s[NE];
  for (int i = 0; i < NE; ++i) s[i] = 1.f / (1.f + __expf(-logits[b * NE + i]));
  float gs[4];
  for (int g = 0; g < 4; ++g) {
    float m1 = -1e30f, m2 = -1e30f;
    for (int j = 0; j < 4; ++j) {
      const float v = s[g * 4 + j];
      if (v > m1) { m2 = m1; m1 = v; } else if (v > m2) m2 = v;
    }
    gs[g] = m1 + m2;
  }
  int g1 = 0;
  for (int g = 1; g < 4; ++g) if (gs[g] > gs[g1]) g1 = g;
  int g2 = -1;
  for (int g = 0; g < 4; ++g) { if (g == g1) continue; if (g2 < 0 || gs[g] > gs[g2]) g2 = g; }
  float masked[NE];
  for (int i = 0; i < NE; ++i) { const int g = i >> 2; masked[i] = (g == g1 || g == g2) ? s[i] : 0.f; }
  int idx[4]; float sum = 0.f;
  for (int t = 0; t < 4; ++t) {
    int bi = -1; float bv = -1e30f;
    for (int i = 0; i < NE; ++i) {
      bool used = false;
      for (int q = 0; q < t; ++q) if (idx[q] == i) used = true;
      if (!used && masked[i] > bv) { bv = masked[i]; bi = i; }
    }
    idx[t] = bi; sum += s[bi];
  }
  const float inv = 1.f / (sum + 1e-20f);  // ROUTE_SCALE = 1
  for (int i = 0; i < NE; ++i) w[b * NE + i] = 0.f;
  for (int t = 0; t < 4; ++t) w[b * NE + idx[t]] = s[idx[t]] * inv;
}

// ---------------- fused MoE expert kernel ----------------
// grid = (B/64, E). Uses CDNA5's 320KB WGP LDS: x tile + full 64x1024 intermediate.
// A-fragments are hoisted to registers (invariant per wave across column tiles).
#define XS_LD 520    // 512 + 8 pad (halfwords)
#define HS_LD 1032   // 1024 + 8 pad
#define MOE_SMEM_BYTES ((64 * XS_LD + 64 * HS_LD) * 2)

__global__ __launch_bounds__(256) void k_moe_fused(
    const unsigned short* __restrict__ Xb,   // [B,512] bf16
    const unsigned short* __restrict__ WgT,  // [E,1024,512] bf16
    const float* __restrict__ bg,            // [E,1024]
    const unsigned short* __restrict__ WuT,  // [E,1024,512]
    const float* __restrict__ bu,            // [E,1024]
    const unsigned short* __restrict__ WdT,  // [E,512,1024]
    const float* __restrict__ bd,            // [E,512]
    const float* __restrict__ route_w,       // [B,E] dense (0 for unselected)
    float* __restrict__ moe_out) {           // [B,512] pre-filled with shared-expert output
  extern __shared__ char smem[];
  unsigned short* xs = (unsigned short*)smem;                    // [64][XS_LD]
  unsigned short* hs = (unsigned short*)smem + 64 * XS_LD;       // [64][HS_LD]
  const int e    = blockIdx.y;
  const int row0 = blockIdx.x * 64;
  const int tid  = threadIdx.x;
  const int lane = tid & 31;
  const int wave = tid >> 5;
  const int wm = wave & 3;
  const int wn = wave >> 2;
  const int nloc = lane & 15;
  const int mofs = (lane >= 16) ? 8 : 0;

  // stage x tile (async global->LDS)
  for (int i = tid; i < 64 * 64; i += 256) {      // 64 rows x (512/8) chunks
    const int row = i >> 6;
    const int kk  = (i & 63) * 8;
    cp16_g2l(xs + row * XS_LD + kk, Xb + (size_t)(row0 + row) * PROJ + kk);
  }
  async_fence();
  __syncthreads();

  const unsigned short* wg = WgT + (size_t)e * HID * PROJ;
  const unsigned short* wu = WuT + (size_t)e * HID * PROJ;
  const float* bge = bg + (size_t)e * HID;
  const float* bue = bu + (size_t)e * HID;

  // phase 1: hs[64,1024] = silu(x@Wg+bg) * (x@Wu+bu)
  {
    v16bf af[16];   // whole K=512 of this wave's 16-row A block, in registers
    const unsigned short* arow = xs + (wm * 16 + nloc) * XS_LD;
#pragma unroll
    for (int ks = 0; ks < 16; ++ks) af[ks] = frag_a(arow + ks * 32, lane);
    for (int j = 0; j < 32; ++j) {
      const int col = wn * 512 + j * 16;
      v8f cg = {}; v8f cu = {};
      const unsigned short* bgp = wg + (size_t)(col + nloc) * PROJ;
      const unsigned short* bup = wu + (size_t)(col + nloc) * PROJ;
#pragma unroll
      for (int ks = 0; ks < 16; ++ks) {
        cg = wmma_bf16(af[ks], frag_b(bgp + ks * 32, lane), cg);
        cu = wmma_bf16(af[ks], frag_b(bup + ks * 32, lane), cu);
      }
      const int c = col + nloc;
      const float bgv = bge[c], buv = bue[c];
#pragma unroll
      for (int i = 0; i < 8; ++i) {
        const int m = wm * 16 + i + mofs;
        const float g = cg[i] + bgv;
        const float u = cu[i] + buv;
        hs[m * HS_LD + c] = f32_to_bf16(silu_f(g) * u);
      }
    }
  }
  __syncthreads();

  // phase 2: y[64,512] = hs @ Wd + bd ; moe_out += route_w[:,e] * y
  {
    const unsigned short* wd = WdT + (size_t)e * PROJ * HID;
    const float* bde = bd + (size_t)e * PROJ;
    v16bf af[32];   // whole K=1024 of this wave's 16-row intermediate block
    const unsigned short* arow = hs + (wm * 16 + nloc) * HS_LD;
#pragma unroll
    for (int ks = 0; ks < 32; ++ks) af[ks] = frag_a(arow + ks * 32, lane);
    for (int j = 0; j < 16; ++j) {
      const int col = wn * 256 + j * 16;
      v8f c2 = {};
      const unsigned short* bdp = wd + (size_t)(col + nloc) * HID;
#pragma unroll
      for (int ks = 0; ks < 32; ++ks)
        c2 = wmma_bf16(af[ks], frag_b(bdp + ks * 32, lane), c2);
      const int c = col + nloc;
      const float bdv = bde[c];
#pragma unroll
      for (int i = 0; i < 8; ++i) {
        const int r = row0 + wm * 16 + i + mofs;
        const float wgt = route_w[(size_t)r * NE + e];
        if (wgt != 0.f)
          unsafeAtomicAdd(&moe_out[(size_t)r * PROJ + c], (c2[i] + bdv) * wgt);
      }
    }
  }
}

// ---------------- host side ----------------
static inline void launch_gemm(hipStream_t s, const GemmParams& p) {
  dim3 grid((unsigned)((p.N + 127) / 128), (unsigned)((p.M + 127) / 128), 1);
  k_gemm<<<grid, dim3(256), 0, s>>>(p);
}
static GemmParams g_dense(const void* A, int adt, long lda, const unsigned short* Bt,
                          const float* bias, void* O, int odt, long ldo,
                          int M, int N, int K, int ep, const unsigned short* aux) {
  GemmParams p{}; p.A = A; p.a_dtype = adt; p.a_mode = 0; p.lda = lda; p.Bt = Bt;
  p.bias = bias; p.aux = aux; p.O = O; p.o_dtype = odt; p.o_mode = 0; p.ldo = ldo;
  p.M = M; p.N = N; p.K = K; p.ep = ep; return p;
}
static GemmParams g_conv(const void* A, int amode, int cin, int tin, int tout, int cstride,
                         const unsigned short* Bt, const float* bias, void* O,
                         int M, int N, int K) {
  GemmParams p{}; p.A = A; p.a_dtype = (amode == 1) ? 0 : 1; p.a_mode = amode;
  p.cin = cin; p.tin = tin; p.tout = tout; p.cstride = cstride; p.Bt = Bt; p.bias = bias;
  p.O = O; p.o_dtype = 1; p.o_mode = 1; p.M = M; p.N = N; p.K = K; p.ep = 1; return p;
}

extern "C" void kernel_launch(void* const* d_in, const int* in_sizes, int n_in,
                              void* d_out, int out_size, void* d_ws, size_t ws_size,
                              hipStream_t stream) {
  (void)in_sizes; (void)n_in; (void)out_size; (void)ws_size;
  // inputs (setup_inputs order)
  const float* cur_obs   = (const float*)d_in[0];
  const float* cur_hist  = (const float*)d_in[1];   // [B,25,96]
  const float* fut_ref   = (const float*)d_in[2];
  const float* vae_noise = (const float*)d_in[3];
  const float* conv1_w = (const float*)d_in[4];  const float* conv1_b = (const float*)d_in[5];
  const float* conv2_w = (const float*)d_in[6];  const float* conv2_b = (const float*)d_in[7];
  const float* conv3_w = (const float*)d_in[8];  const float* conv3_b = (const float*)d_in[9];
  const float* hlin1_w = (const float*)d_in[10]; const float* hlin1_b = (const float*)d_in[11];
  const float* hlin2_w = (const float*)d_in[12]; const float* hlin2_b = (const float*)d_in[13];
  const float* enc1_w  = (const float*)d_in[14]; const float* enc1_b  = (const float*)d_in[15];
  const float* enc2_w  = (const float*)d_in[16]; const float* enc2_b  = (const float*)d_in[17];
  const float* enc3_w  = (const float*)d_in[18]; const float* enc3_b  = (const float*)d_in[19];
  const float* proj_w  = (const float*)d_in[20]; const float* proj_b  = (const float*)d_in[21];
  const float* router_w = (const float*)d_in[22]; const float* router_b = (const float*)d_in[23];
  const float* Wg = (const float*)d_in[24]; const float* bg = (const float*)d_in[25];
  const float* Wu = (const float*)d_in[26]; const float* bu = (const float*)d_in[27];
  const float* Wd = (const float*)d_in[28]; const float* bd = (const float*)d_in[29];
  const float* sh_g_w = (const float*)d_in[30]; const float* sh_g_b = (const float*)d_in[31];
  const float* sh_u_w = (const float*)d_in[32]; const float* sh_u_b = (const float*)d_in[33];
  const float* sh_d_w = (const float*)d_in[34]; const float* sh_d_b = (const float*)d_in[35];
  const float* out1_w = (const float*)d_in[36]; const float* out1_b = (const float*)d_in[37];
  const float* out2_w = (const float*)d_in[38]; const float* out2_b = (const float*)d_in[39];
  const float* head_w = (const float*)d_in[40]; const float* head_b = (const float*)d_in[41];

  // workspace bump allocator (identical layout every call)
  char* ws = (char*)d_ws;
  size_t off = 0;
  auto alloc = [&](size_t bytes) -> void* {
    void* p = ws + off; off += (bytes + 255) & ~(size_t)255; return p;
  };
  typedef unsigned short u16;
  u16* c1w  = (u16*)alloc((size_t)CONV1C * HIST_C * 3 * 2);
  u16* c2w  = (u16*)alloc((size_t)CONV2C * CONV1C * 3 * 2);
  u16* c3w  = (u16*)alloc((size_t)CONV3C * CONV2C * 3 * 2);
  u16* hlin1T = (u16*)alloc((size_t)HID * HID * 2);
  u16* hlin2T = (u16*)alloc((size_t)HLAT * HID * 2);
  u16* enc1T  = (u16*)alloc((size_t)HID * FUT_FLAT * 2);
  u16* enc2T  = (u16*)alloc((size_t)HID * HID * 2);
  u16* enc3T  = (u16*)alloc((size_t)2 * FLATD * HID * 2);
  u16* projT  = (u16*)alloc((size_t)PROJ * 384 * 2);
  u16* routerT= (u16*)alloc((size_t)NE * PROJ * 2);
  u16* WgT = (u16*)alloc((size_t)NE * HID * PROJ * 2);
  u16* WuT = (u16*)alloc((size_t)NE * HID * PROJ * 2);
  u16* WdT = (u16*)alloc((size_t)NE * PROJ * HID * 2);
  u16* shgT = (u16*)alloc((size_t)2048 * PROJ * 2);
  u16* shuT = (u16*)alloc((size_t)2048 * PROJ * 2);
  u16* shdT = (u16*)alloc((size_t)PROJ * 2048 * 2);
  u16* out1T = (u16*)alloc((size_t)HID * PROJ * 2);
  u16* out2T = (u16*)alloc((size_t)HID * HID * 2);
  u16* headT = (u16*)alloc((size_t)OUTD * HID * 2);
  u16* h1 = (u16*)alloc((size_t)BN * CONV1C * 25 * 2);
  u16* h2 = (u16*)alloc((size_t)BN * CONV2C * 13 * 2);
  u16* h3 = (u16*)alloc((size_t)BN * CONV3C * 7 * 2);
  u16* pooled = (u16*)alloc((size_t)BN * CONV3C * 2);
  u16* hl1    = (u16*)alloc((size_t)BN * HID * 2);
  u16* hist_lat = (u16*)alloc((size_t)BN * HLAT * 2);
  float* e3   = (float*)alloc((size_t)BN * 128 * 4);
  u16* acat   = (u16*)alloc((size_t)BN * 384 * 2);
  float* x_f32 = (float*)alloc((size_t)BN * PROJ * 4);
  u16* x_bf    = (u16*)alloc((size_t)BN * PROJ * 2);
  float* logits  = (float*)alloc((size_t)BN * NE * 4);
  float* route_w = (float*)alloc((size_t)BN * NE * 4);
  u16* hsh     = (u16*)alloc((size_t)BN * 2048 * 2);
  float* moe_out = (float*)alloc((size_t)BN * PROJ * 4);
  u16* o2 = (u16*)alloc((size_t)BN * HID * 2);
  // aliases into dead conv buffers
  u16* e1 = h1;   // [B,1024] bf16 (h1 dead after conv2)
  u16* e2 = h2;   // [B,1024] bf16 (h2 dead after conv3)
  u16* o1 = h3;   // [B,1024] bf16 (h3 dead after pool)

  const dim3 cvg(256), cvb(256);
  // ---- weight conversion / transposition ----
  k_convert_bf16<<<cvg, cvb, 0, stream>>>(conv1_w, c1w, (size_t)CONV1C * HIST_C * 3);
  k_convert_bf16<<<cvg, cvb, 0, stream>>>(conv2_w, c2w, (size_t)CONV2C * CONV1C * 3);
  k_convert_bf16<<<cvg, cvb, 0, stream>>>(conv3_w, c3w, (size_t)CONV3C * CONV2C * 3);
  k_transpose_bf16<<<dim3(256, 1), cvb, 0, stream>>>(hlin1_w, hlin1T, HID, HID);
  k_transpose_bf16<<<dim3(256, 1), cvb, 0, stream>>>(hlin2_w, hlin2T, HID, HLAT);
  k_transpose_bf16<<<dim3(512, 1), cvb, 0, stream>>>(enc1_w, enc1T, FUT_FLAT, HID);
  k_transpose_bf16<<<dim3(256, 1), cvb, 0, stream>>>(enc2_w, enc2T, HID, HID);
  k_transpose_bf16<<<dim3(128, 1), cvb, 0, stream>>>(enc3_w, enc3T, HID, 2 * FLATD);
  k_transpose_bf16<<<dim3(128, 1), cvb, 0, stream>>>(proj_w, projT, 384, PROJ);
  k_transpose_bf16<<<dim3(32, 1),  cvb, 0, stream>>>(router_w, routerT, PROJ, NE);
  k_transpose_bf16<<<dim3(256, NE), cvb, 0, stream>>>(Wg, WgT, PROJ, HID);
  k_transpose_bf16<<<dim3(256, NE), cvb, 0, stream>>>(Wu, WuT, PROJ, HID);
  k_transpose_bf16<<<dim3(256, NE), cvb, 0, stream>>>(Wd, WdT, HID, PROJ);
  k_transpose_bf16<<<dim3(256, 1), cvb, 0, stream>>>(sh_g_w, shgT, PROJ, 2048);
  k_transpose_bf16<<<dim3(256, 1), cvb, 0, stream>>>(sh_u_w, shuT, PROJ, 2048);
  k_transpose_bf16<<<dim3(256, 1), cvb, 0, stream>>>(sh_d_w, shdT, 2048, PROJ);
  k_transpose_bf16<<<dim3(256, 1), cvb, 0, stream>>>(out1_w, out1T, PROJ, HID);
  k_transpose_bf16<<<dim3(256, 1), cvb, 0, stream>>>(out2_w, out2T, HID, HID);
  k_transpose_bf16<<<dim3(32, 1),  cvb, 0, stream>>>(head_w, headT, HID, OUTD);

  // ---- history conv encoder (im2col WMMA GEMMs, SiLU fused, CHW bf16 out) ----
  launch_gemm(stream, g_conv(cur_hist, 1, HIST_C, 25, 25, 1, c1w, conv1_b, h1,
                             BN * 25, CONV1C, HIST_C * 3));
  launch_gemm(stream, g_conv(h1, 2, CONV1C, 25, 13, 2, c2w, conv2_b, h2,
                             BN * 13, CONV2C, CONV1C * 3));
  launch_gemm(stream, g_conv(h2, 2, CONV2C, 13, 7, 2, c3w, conv3_b, h3,
                             BN * 7, CONV3C, CONV2C * 3));
  k_meanpool7<<<dim3(1024), cvb, 0, stream>>>(h3, pooled);
  launch_gemm(stream, g_dense(pooled, 1, HID, hlin1T, hlin1_b, hl1, 1, HID,
                              BN, HID, HID, 1, nullptr));
  launch_gemm(stream, g_dense(hl1, 1, HID, hlin2T, hlin2_b, hist_lat, 1, HLAT,
                              BN, HLAT, HID, 0, nullptr));

  // ---- fut-ref VAE encoder ----
  launch_gemm(stream, g_dense(fut_ref, 0, FUT_FLAT, enc1T, enc1_b, e1, 1, HID,
                              BN, HID, FUT_FLAT, 1, nullptr));
  launch_gemm(stream, g_dense(e1, 1, HID, enc2T, enc2_b, e2, 1, HID,
                              BN, HID, HID, 1, nullptr));
  launch_gemm(stream, g_dense(e2, 1, HID, enc3T, enc3_b, e3, 0, 128,
                              BN, 2 * FLATD, HID, 0, nullptr));

  // ---- reparameterize + concat + projection ----
  k_reparam_concat<<<dim3(1024), cvb, 0, stream>>>(cur_obs, hist_lat, e3, vae_noise, acat);
  launch_gemm(stream, g_dense(acat, 1, 384, projT, proj_b, x_f32, 0, PROJ,
                              BN, PROJ, 384, 0, nullptr));
  k_convert_bf16<<<dim3(1024), cvb, 0, stream>>>(x_f32, x_bf, (size_t)BN * PROJ);

  // ---- router ----
  launch_gemm(stream, g_dense(x_f32, 0, PROJ, routerT, router_b, logits, 0, NE,
                              BN, NE, PROJ, 0, nullptr));
  k_router_topk<<<dim3((BN + 255) / 256), cvb, 0, stream>>>(logits, route_w);

  // ---- shared experts (writes moe_out base) ----
  launch_gemm(stream, g_dense(x_bf, 1, PROJ, shgT, sh_g_b, hsh, 1, 2048,
                              BN, 2048, PROJ, 1, nullptr));
  launch_gemm(stream, g_dense(x_bf, 1, PROJ, shuT, sh_u_b, hsh, 1, 2048,
                              BN, 2048, PROJ, 2, hsh));
  launch_gemm(stream, g_dense(hsh, 1, 2048, shdT, sh_d_b, moe_out, 0, PROJ,
                              BN, PROJ, 2048, 0, nullptr));

  // ---- fused MoE experts (atomic combine on top of shared output) ----
  (void)hipFuncSetAttribute((const void*)k_moe_fused,
                            hipFuncAttributeMaxDynamicSharedMemorySize, MOE_SMEM_BYTES);
  k_moe_fused<<<dim3(BN / 64, NE), dim3(256), MOE_SMEM_BYTES, stream>>>(
      x_bf, WgT, bg, WuT, bu, WdT, bd, route_w, moe_out);

  // ---- output MLP + head ----
  launch_gemm(stream, g_dense(moe_out, 0, PROJ, out1T, out1_b, o1, 1, HID,
                              BN, HID, PROJ, 1, nullptr));
  launch_gemm(stream, g_dense(o1, 1, HID, out2T, out2_b, o2, 1, HID,
                              BN, HID, HID, 0, nullptr));
  launch_gemm(stream, g_dense(o2, 1, HID, headT, head_b, d_out, 0, OUTD,
                              BN, OUTD, HID, 0, nullptr));
}
